// Transformer_59279138619940
// MI455X (gfx1250) — matrix operand order repaired and checked by reading, not compiled, not run on hardware
//
#include <hip/hip_runtime.h>

// ---------------------------------------------------------------------------
// Types
// ---------------------------------------------------------------------------
typedef __bf16 bf16_t;
typedef __attribute__((ext_vector_type(16))) __bf16 v16bf;
typedef __attribute__((ext_vector_type(8)))  float  v8f;
typedef __attribute__((ext_vector_type(4)))  unsigned u32x4;
typedef __attribute__((ext_vector_type(4)))  int      i32x4;
typedef __attribute__((ext_vector_type(8)))  int      i32x8;
struct U32x8 { u32x4 lo, hi; };

#if defined(__has_builtin)
#if __has_builtin(__builtin_amdgcn_tensor_load_to_lds) && __has_builtin(__builtin_amdgcn_s_wait_tensorcnt)
#define HAVE_TDM 1
#endif
#endif
#ifndef HAVE_TDM
#define HAVE_TDM 0
#endif

// Model dims
#define D_   512
#define S_   512
#define H_   8
#define HD_  64
#define B_   8
#define L_   4
#define V_   32000
#define FF_  2048
#define MT_  4096          // B_*S_ flattened tokens
#define KIN_ 144
#define KPAD_ 160
#define BK_  64            // K-tile per pipeline stage

__device__ __forceinline__ bf16_t f2bf(float f) {
  unsigned u = __builtin_bit_cast(unsigned, f);
  unsigned r = u + 0x7FFFu + ((u >> 16) & 1u);   // round-to-nearest-even
  unsigned short h = (unsigned short)(r >> 16);
  return __builtin_bit_cast(bf16_t, h);
}

// ---------------------------------------------------------------------------
// WMMA fragment loaders from LDS tiles (tile layout: [rows][64] bf16, K-major).
// A-matrix 16x32 bf16 (ISA 7.12.2): lane<16 row=lane, K chunks {0..7},{16..23};
// lane>=16 row=lane-16, chunks {8..15},{24..31}. kq selects 32-col window.
// ---------------------------------------------------------------------------
__device__ __forceinline__ v16bf frag_a64(const bf16_t* base, int row0, int kq, int lane) {
  int r  = row0 + (lane & 15);
  int kb = kq * 32 + ((lane >> 4) << 3);          // +0 or +8
  const bf16_t* p = base + r * BK_;
  U32x8 u;
  u.lo = *(const u32x4*)(p + kb);
  u.hi = *(const u32x4*)(p + kb + 16);
  return __builtin_bit_cast(v16bf, u);
}
// B-matrix 32x16 bf16: lane<16 col=lane holds K chunks {0..7},{16..23}; lane>=16
// col=lane-16 holds chunks {8..15},{24..31}.
__device__ __forceinline__ v16bf frag_b64(const bf16_t* base, int row0, int kq, int lane) {
  int r  = row0 + (lane & 15);
  int kb = kq * 32 + ((lane >> 4) << 3);          // +0 or +8
  const bf16_t* p = base + r * BK_ + kb;
  U32x8 u;
  u.lo = *(const u32x4*)(p);
  u.hi = *(const u32x4*)(p + 16);
  return __builtin_bit_cast(v16bf, u);
}

#if HAVE_TDM
// ---------------------------------------------------------------------------
// Tensor Data Mover: async 2D tile load global->LDS (D# per ISA ch.8).
// data_size=2B; tensor_dim0/1 = remaining cols/rows (OOB reads return zero);
// group0: count=1, lds_addr, global_addr, type=2.
// ---------------------------------------------------------------------------
__device__ __forceinline__ void tdm_load_2d(const void* gaddr, unsigned lds_byte_off,
                                            unsigned rem_cols, unsigned rem_rows,
                                            unsigned tile_cols, unsigned tile_rows,
                                            unsigned long long stride_elems) {
  unsigned long long ga = (unsigned long long)gaddr;
  u32x4 g0;
  g0[0] = 1u;                                              // count=1, user mode
  g0[1] = lds_byte_off;                                    // lds_addr (bytes)
  g0[2] = (unsigned)(ga & 0xFFFFFFFFu);                    // global_addr[31:0]
  g0[3] = (unsigned)((ga >> 32) & 0x1FFFFFFu) | (2u << 30);// addr[56:32] | type=2
  i32x8 g1;
  g1[0] = 0x00010000;                                      // data_size=1 (2 bytes)
  g1[1] = (int)((rem_cols & 0xFFFFu) << 16);               // tensor_dim0[15:0]
  g1[2] = (int)((rem_cols >> 16) | ((rem_rows & 0xFFFFu) << 16));
  g1[3] = (int)((rem_rows >> 16) | (tile_cols << 16));     // tile_dim0
  g1[4] = (int)tile_rows;                                  // tile_dim1 (dim2=0)
  g1[5] = (int)(stride_elems & 0xFFFFFFFFu);               // dim0_stride[31:0]
  g1[6] = (int)((stride_elems >> 32) & 0xFFFFu);           // dim0_stride[47:32]
  g1[7] = 0;
  i32x4 z4 = {0, 0, 0, 0};
#if defined(__clang_major__) && __clang_major__ >= 23
  i32x8 z8 = {0, 0, 0, 0, 0, 0, 0, 0};
  __builtin_amdgcn_tensor_load_to_lds(g0, g1, z4, z4, z8, 0);
#else
  __builtin_amdgcn_tensor_load_to_lds(g0, g1, z4, z4, 0);
#endif
}
#endif

// Manual fallback staging: 128x64 bf16 tile, whole block cooperates.
__device__ __forceinline__ void stage_manual(const bf16_t* __restrict__ G, long long ld,
                                             int r0, int rmax, int k0, int K,
                                             bf16_t* __restrict__ dst, int tid) {
#pragma unroll
  for (int i = 0; i < 4; i++) {
    int idx = tid + i * 256;
    int row = idx >> 3, seg = idx & 7;
    u32x4 v = {0u, 0u, 0u, 0u};
    int gr = r0 + row, gc = k0 + seg * 8;
    if (gr < rmax && gc < K) v = *(const u32x4*)(G + (long long)gr * ld + gc);
    *(u32x4*)(dst + row * BK_ + seg * 8) = v;
  }
}

// ---------------------------------------------------------------------------
// Generic batched GEMM:  C[M,N] = A[M,K] * Bt[N,K]^T  (+bias, +relu)
// Batch z -> (b=z/Hdiv, h=z%Hdiv); offsets = b*s?b + h*s?h.
// Tile 128x128xBK_(64), 256 threads = 8 wave32 (2x4), wave tile 64x32.
// Double-buffered LDS fed by TDM async tile loads (s_wait_tensorcnt pipeline).
// ---------------------------------------------------------------------------
__global__ __launch_bounds__(256)
void k_gemm(const bf16_t* __restrict__ A,  long long lda, long long sAb, long long sAh,
            const bf16_t* __restrict__ Bt, long long ldb, long long sBb, long long sBh,
            float* __restrict__ Cf, bf16_t* __restrict__ Cb,
            long long ldc, long long sCb, long long sCh,
            int M, int N, int K,
            const float* __restrict__ bias, int relu, int Hdiv) {
  __shared__ __align__(16) bf16_t sA[2][128 * BK_];
  __shared__ __align__(16) bf16_t sB[2][128 * BK_];

  const int tid  = threadIdx.x;
  const int lane = tid & 31;
  const int wave = tid >> 5;
  const int wm   = wave >> 2;              // 0..1
  const int wn   = wave & 3;               // 0..3
  const int m0   = blockIdx.y * 128;
  const int n0   = blockIdx.x * 128;
  const int z    = blockIdx.z;
  const int bz   = z / Hdiv;
  const int hz   = z % Hdiv;

  const bf16_t* Ab = A  + bz * sAb + hz * sAh;
  const bf16_t* Bb = Bt + bz * sBb + hz * sBh;
  const long long offC = bz * sCb + hz * sCh;

  v8f acc[4][2];
#pragma unroll
  for (int a = 0; a < 4; a++)
#pragma unroll
    for (int b = 0; b < 2; b++)
#pragma unroll
      for (int j = 0; j < 8; j++) acc[a][b][j] = 0.f;

#if HAVE_TDM
  // Each wave DMAs its own 16-row slice of the A and B tiles (2 TDM ops/wave).
  auto issue = [&](int buf, int k0) {
    int ra  = m0 + wave * 16;
    int rac = (ra < M) ? ra : (M - 1);
    unsigned remRA = (ra < M) ? (unsigned)(M - ra) : 0u;
    tdm_load_2d(Ab + (long long)rac * lda + k0,
                (unsigned)(size_t)&sA[buf][wave * 16 * BK_],
                (unsigned)(K - k0), remRA, BK_, 16, (unsigned long long)lda);
    int rb  = n0 + wave * 16;
    int rbc = (rb < N) ? rb : (N - 1);
    unsigned remRB = (rb < N) ? (unsigned)(N - rb) : 0u;
    tdm_load_2d(Bb + (long long)rbc * ldb + k0,
                (unsigned)(size_t)&sB[buf][wave * 16 * BK_],
                (unsigned)(K - k0), remRB, BK_, 16, (unsigned long long)ldb);
  };
  issue(0, 0);
#else
  stage_manual(Ab, lda, m0, M, 0, K, sA[0], tid);
  stage_manual(Bb, ldb, n0, N, 0, K, sB[0], tid);
#endif

  int buf = 0;
  for (int k0 = 0; k0 < K; k0 += BK_) {
    const bool more = (k0 + BK_ < K);
    if (more) {
#if HAVE_TDM
      issue(buf ^ 1, k0 + BK_);
#else
      stage_manual(Ab, lda, m0, M, k0 + BK_, K, sA[buf ^ 1], tid);
      stage_manual(Bb, ldb, n0, N, k0 + BK_, K, sB[buf ^ 1], tid);
#endif
    }
#if HAVE_TDM
    // Tensor ops complete in order per wave: cnt<=2 => current buffer landed.
    if (more) __builtin_amdgcn_s_wait_tensorcnt(2);
    else      __builtin_amdgcn_s_wait_tensorcnt(0);
#endif
    __syncthreads();                        // all waves' slices visible

    const bf16_t* cA = sA[buf];
    const bf16_t* cB = sB[buf];
#pragma unroll
    for (int kq = 0; kq < 2; kq++) {
      v16bf af[4], bfr[2];
#pragma unroll
      for (int mt = 0; mt < 4; mt++) af[mt] = frag_a64(cA, wm * 64 + mt * 16, kq, lane);
#pragma unroll
      for (int nt = 0; nt < 2; nt++) bfr[nt] = frag_b64(cB, wn * 32 + nt * 16, kq, lane);
#pragma unroll
      for (int mt = 0; mt < 4; mt++)
#pragma unroll
        for (int nt = 0; nt < 2; nt++)
          acc[mt][nt] = __builtin_amdgcn_wmma_f32_16x16x32_bf16(
              false, af[mt], false, bfr[nt], (short)0, acc[mt][nt], false, false);
    }
    __syncthreads();                        // done reading buf before overwrite
    buf ^= 1;
  }

  // Epilogue: C/D layout — VGPR j: M=row0+j (lanes 0-15) / row0+8+j (16-31)
#pragma unroll
  for (int mt = 0; mt < 4; mt++) {
    int r0 = m0 + wm * 64 + mt * 16 + ((lane >> 4) << 3);
#pragma unroll
    for (int nt = 0; nt < 2; nt++) {
      int c = n0 + wn * 32 + nt * 16 + (lane & 15);
      if (c < N) {
        float bv = bias ? bias[c] : 0.f;
#pragma unroll
        for (int j = 0; j < 8; j++) {
          int r = r0 + j;
          if (r < M) {
            float v = acc[mt][nt][j] + bv;
            if (relu) v = fmaxf(v, 0.f);
            long long o = offC + (long long)r * ldc + c;
            if (Cf) Cf[o] = v;
            if (Cb) Cb[o] = f2bf(v);
          }
        }
      }
    }
  }
}

// ---------------------------------------------------------------------------
// Weight prep: fp32 [K,N] -> bf16 [N,Kp] transpose+convert (+zero-pad K->Kp)
// ---------------------------------------------------------------------------
__global__ void k_transpose_cvt(const float* __restrict__ in, bf16_t* __restrict__ out,
                                int K, int N, int Kp) {
  __shared__ float tile[32][33];
  int kb = blockIdx.y * 32, nb = blockIdx.x * 32;
  int tx = threadIdx.x, ty = threadIdx.y;        // 32x8
#pragma unroll
  for (int i = 0; i < 32; i += 8) {
    int k = kb + ty + i, n = nb + tx;
    tile[ty + i][tx] = (k < K && n < N) ? in[(long long)k * N + n] : 0.f;
  }
  __syncthreads();
#pragma unroll
  for (int i = 0; i < 32; i += 8) {
    int n = nb + ty + i, k = kb + tx;
    if (n < N && k < Kp) out[(long long)n * Kp + k] = f2bf(tile[tx][ty + i]);
  }
}

// x [4096,144] f32 -> [4096,160] bf16 zero-padded
__global__ void k_pad_x(const float* __restrict__ x, bf16_t* __restrict__ out) {
  int i = blockIdx.x * 256 + threadIdx.x;
  if (i >= MT_ * KPAD_) return;
  int k = i % KPAD_, m = i / KPAD_;
  out[i] = (k < KIN_) ? f2bf(x[(long long)m * KIN_ + k]) : f2bf(0.f);
}

// xe += pos_embed (broadcast over batch); also emit bf16
__global__ void k_add_pos(float* __restrict__ xe, const float* __restrict__ pos,
                          bf16_t* __restrict__ xb) {
  long long i = (long long)blockIdx.x * 256 + threadIdx.x;
  if (i >= (long long)MT_ * D_) return;
  float v = xe[i] + pos[i % ((long long)S_ * D_)];
  xe[i] = v;
  xb[i] = f2bf(v);
}

// target embedding gather + sinusoidal PE (exponent 2*d/D, even=sin, odd=cos)
__global__ void k_embed_tgt(const int* __restrict__ y, const float* __restrict__ emb,
                            float* __restrict__ hf, bf16_t* __restrict__ hb) {
  long long i = (long long)blockIdx.x * 256 + threadIdx.x;
  if (i >= (long long)MT_ * D_) return;
  int d = (int)(i & (D_ - 1));
  long long bs = i >> 9;                   // b*512 + s
  int s = (int)(bs & (S_ - 1));
  int tok = y[bs];
  float ang = (float)s * __powf(10000.f, -2.f * (float)d / (float)D_);
  float pe  = (d & 1) ? __cosf(ang) : __sinf(ang);
  float v   = emb[(long long)tok * D_ + d] + pe;
  hf[i] = v;
  hb[i] = f2bf(v);
}

// v [B,S,D] bf16 -> vT [B,H,HD,S] bf16
__global__ void k_transpose_heads(const bf16_t* __restrict__ in, bf16_t* __restrict__ out) {
  long long i = (long long)blockIdx.x * 256 + threadIdx.x;
  if (i >= (long long)B_ * S_ * D_) return;
  int d = (int)(i & (D_ - 1));
  long long bs = i >> 9;
  int s = (int)(bs & (S_ - 1));
  int b = (int)(bs >> 9);
  int h = d >> 6, dd = d & 63;
  out[(((long long)(b * H_ + h) * HD_ + dd) << 9) + s] = in[i];
}

// out = LayerNorm(x + res) * g + b ; writes fp32 + bf16. One block per row.
__global__ void k_add_ln(const float* __restrict__ x, const float* __restrict__ r,
                         const float* __restrict__ g, const float* __restrict__ b,
                         float* __restrict__ of, bf16_t* __restrict__ ob) {
  __shared__ float red[128];
  int row = blockIdx.x, t = threadIdx.x;
  const float* xp = x + (long long)row * D_;
  const float* rp = r + (long long)row * D_;
  float v[4]; float s = 0.f;
#pragma unroll
  for (int i = 0; i < 4; i++) { int c = t + i * 128; v[i] = xp[c] + rp[c]; s += v[i]; }
  red[t] = s; __syncthreads();
  for (int st = 64; st > 0; st >>= 1) { if (t < st) red[t] += red[t + st]; __syncthreads(); }
  float mean = red[0] * (1.f / D_); __syncthreads();
  float vs = 0.f;
#pragma unroll
  for (int i = 0; i < 4; i++) { float d = v[i] - mean; vs += d * d; }
  red[t] = vs; __syncthreads();
  for (int st = 64; st > 0; st >>= 1) { if (t < st) red[t] += red[t + st]; __syncthreads(); }
  float inv = rsqrtf(red[0] * (1.f / D_) + 1e-5f);
#pragma unroll
  for (int i = 0; i < 4; i++) {
    int c = t + i * 128;
    float o = (v[i] - mean) * inv * g[c] + b[c];
    long long idx = (long long)row * D_ + c;
    of[idx] = o;
    ob[idx] = f2bf(o);
  }
}

// softmax over Sk=512 with optional (causal | pad) mask; writes bf16 att.
// row = (b*H + h)*S + q ; scale = 1/sqrt(D)
__global__ void k_softmax(const float* __restrict__ sc, bf16_t* __restrict__ att,
                          const int* __restrict__ y, int causal, float scale) {
  __shared__ float red[128];
  int row = blockIdx.x, t = threadIdx.x;
  int q = row & (S_ - 1);
  int b = row >> 12;                        // row / (H_*S_)
  const float* p = sc + (long long)row * S_;
  float v[4]; float mx = -3.4e38f;
#pragma unroll
  for (int i = 0; i < 4; i++) {
    int k = t + i * 128;
    float s = p[k];
    if (causal && ((k > q) || (y[(b << 9) + k] == 0))) s -= 1e9f;
    s *= scale;
    v[i] = s;
    mx = fmaxf(mx, s);
  }
  red[t] = mx; __syncthreads();
  for (int st = 64; st > 0; st >>= 1) { if (t < st) red[t] = fmaxf(red[t], red[t + st]); __syncthreads(); }
  mx = red[0]; __syncthreads();
  float sum = 0.f;
#pragma unroll
  for (int i = 0; i < 4; i++) { v[i] = __expf(v[i] - mx); sum += v[i]; }
  red[t] = sum; __syncthreads();
  for (int st = 64; st > 0; st >>= 1) { if (t < st) red[t] += red[t + st]; __syncthreads(); }
  float inv = 1.f / red[0];
#pragma unroll
  for (int i = 0; i < 4; i++)
    att[(long long)row * S_ + t + i * 128] = f2bf(v[i] * inv);
}

// ---------------------------------------------------------------------------
// Host orchestration
// ---------------------------------------------------------------------------
typedef long long ll;

static void launch_gemm(hipStream_t st,
                        const bf16_t* A, ll lda, ll sAb, ll sAh,
                        const bf16_t* Bt, ll ldb, ll sBb, ll sBh,
                        float* Cf, bf16_t* Cb, ll ldc, ll sCb, ll sCh,
                        int M, int N, int K, const float* bias, int relu,
                        int batch, int Hdiv) {
  dim3 grid((N + 127) / 128, (M + 127) / 128, batch);
  k_gemm<<<grid, 256, 0, st>>>(A, lda, sAb, sAh, Bt, ldb, sBb, sBh,
                               Cf, Cb, ldc, sCb, sCh, M, N, K, bias, relu, Hdiv);
}

extern "C" void kernel_launch(void* const* d_in, const int* in_sizes, int n_in,
                              void* d_out, int out_size, void* d_ws, size_t ws_size,
                              hipStream_t stream) {
  // ---- inputs (setup_inputs order) ----
  const float* x_in      = (const float*)d_in[0];
  const int*   y_in      = (const int*)  d_in[1];
  const float* fc_w      = (const float*)d_in[2];
  const float* fc_b      = (const float*)d_in[3];
  const float* pos_emb   = (const float*)d_in[4];
  const float* emb_tab   = (const float*)d_in[5];
  const float* eaw       = (const float*)d_in[6];
  const float* eab       = (const float*)d_in[7];
  const float* elnw      = (const float*)d_in[8];
  const float* elnb      = (const float*)d_in[9];
  const float* ef1w      = (const float*)d_in[10];
  const float* ef1b      = (const float*)d_in[11];
  const float* ef2w      = (const float*)d_in[12];
  const float* ef2b      = (const float*)d_in[13];
  const float* dnw       = (const float*)d_in[14];
  const float* dnb       = (const float*)d_in[15];
  const float* dsw       = (const float*)d_in[16];
  const float* dsb       = (const float*)d_in[17];
  const float* dcw       = (const float*)d_in[18];
  const float* dcb       = (const float*)d_in[19];
  const float* dlnw      = (const float*)d_in[20];
  const float* dlnb      = (const float*)d_in[21];
  const float* df1w      = (const float*)d_in[22];
  const float* df1b      = (const float*)d_in[23];
  const float* df2w      = (const float*)d_in[24];
  const float* df2b      = (const float*)d_in[25];
  const float* out1_w    = (const float*)d_in[26];
  const float* out1_b    = (const float*)d_in[27];
  const float* out2_w    = (const float*)d_in[28];
  const float* out2_b    = (const float*)d_in[29];
  float* logits = (float*)d_out;

  // ---- workspace bump allocator ----
  char* base = (char*)d_ws;
  size_t off = 0;
  auto alloc = [&](size_t bytes) -> void* {
    void* p = base + off;
    off += (bytes + 255) & ~(size_t)255;
    return p;
  };
  const size_t DD = (size_t)D_ * D_;
  // bf16 transposed weights [N,K]
  bf16_t* fcT    = (bf16_t*)alloc((size_t)D_ * KPAD_ * 2);
  bf16_t* encAT  = (bf16_t*)alloc(4 * DD * 2);
  bf16_t* encF1T = (bf16_t*)alloc((size_t)FF_ * D_ * 2);
  bf16_t* encF2T = (bf16_t*)alloc((size_t)D_ * FF_ * 2);
  bf16_t* decST  = (bf16_t*)alloc((size_t)L_ * 4 * DD * 2);
  bf16_t* decCT  = (bf16_t*)alloc((size_t)L_ * 4 * DD * 2);
  bf16_t* decF1T = (bf16_t*)alloc((size_t)L_ * FF_ * D_ * 2);
  bf16_t* decF2T = (bf16_t*)alloc((size_t)L_ * D_ * FF_ * 2);
  bf16_t* out1T  = (bf16_t*)alloc(DD * 2);
  bf16_t* out2T  = (bf16_t*)alloc((size_t)V_ * D_ * 2);
  // activations
  bf16_t* xpad   = (bf16_t*)alloc((size_t)MT_ * KPAD_ * 2);
  float*  xe_f   = (float*) alloc((size_t)MT_ * D_ * 4);
  bf16_t* xe_b   = (bf16_t*)alloc((size_t)MT_ * D_ * 2);
  float*  enc_f  = (float*) alloc((size_t)MT_ * D_ * 4);
  bf16_t* enc_b  = (bf16_t*)alloc((size_t)MT_ * D_ * 2);
  float*  h_f    = (float*) alloc((size_t)MT_ * D_ * 4);
  bf16_t* h_b    = (bf16_t*)alloc((size_t)MT_ * D_ * 2);
  float*  qy_f   = (float*) alloc((size_t)MT_ * D_ * 4);
  bf16_t* qy_b   = (bf16_t*)alloc((size_t)MT_ * D_ * 2);
  float*  x1_f   = (float*) alloc((size_t)MT_ * D_ * 4);
  bf16_t* x1_b   = (bf16_t*)alloc((size_t)MT_ * D_ * 2);
  float*  bufT   = (float*) alloc((size_t)MT_ * D_ * 4);
  bf16_t* q_b    = (bf16_t*)alloc((size_t)MT_ * D_ * 2);
  bf16_t* k_b    = (bf16_t*)alloc((size_t)MT_ * D_ * 2);
  bf16_t* v_b    = (bf16_t*)alloc((size_t)MT_ * D_ * 2);
  bf16_t* vT_b   = (bf16_t*)alloc((size_t)MT_ * D_ * 2);
  float*  scores = (float*) alloc((size_t)B_ * H_ * S_ * S_ * 4);
  bf16_t* att_b  = (bf16_t*)alloc((size_t)B_ * H_ * S_ * S_ * 2);
  bf16_t* ctx_b  = (bf16_t*)alloc((size_t)MT_ * D_ * 2);
  bf16_t* ff1_b  = (bf16_t*)alloc((size_t)MT_ * FF_ * 2);
  bf16_t* h1_b   = (bf16_t*)alloc((size_t)MT_ * D_ * 2);
  (void)ws_size; (void)in_sizes; (void)n_in; (void)out_size;

  auto trs = [&](const float* in, bf16_t* out, int K, int N, int Kp) {
    dim3 g((N + 31) / 32, (Kp + 31) / 32);
    k_transpose_cvt<<<g, dim3(32, 8), 0, stream>>>(in, out, K, N, Kp);
  };

  // ---- phase 0: weight transposition to bf16 [N,K] ----
  trs(fc_w, fcT, KIN_, D_, KPAD_);
  for (int j = 0; j < 4; j++) trs(eaw + j * DD, encAT + j * DD, D_, D_, D_);
  trs(ef1w, encF1T, D_, FF_, D_);
  trs(ef2w, encF2T, FF_, D_, FF_);
  for (int l = 0; l < L_; l++)
    for (int j = 0; j < 4; j++) {
      trs(dsw + (l * 4 + j) * DD, decST + (l * 4 + j) * DD, D_, D_, D_);
      trs(dcw + (l * 4 + j) * DD, decCT + (l * 4 + j) * DD, D_, D_, D_);
    }
  for (int l = 0; l < L_; l++) {
    trs(df1w + (size_t)l * D_ * FF_, decF1T + (size_t)l * FF_ * D_, D_, FF_, D_);
    trs(df2w + (size_t)l * FF_ * D_, decF2T + (size_t)l * D_ * FF_, FF_, D_, FF_);
  }
  trs(out1_w, out1T, D_, D_, D_);
  trs(out2_w, out2T, D_, V_, D_);

  const int eg = (int)(((size_t)MT_ * D_ + 255) / 256);
  const float inv_sqrt_d = 0.044194173824159216f;   // 1/sqrt(512)

  // full attention sub-block: proj -> scores -> softmax -> ctx -> O-proj -> add+LN
  auto attention = [&](const bf16_t* qsrc, const bf16_t* kvsrc,
                       const bf16_t* wT, const float* bvec, int causal,
                       const float* resid, const float* lnw_, const float* lnb_,
                       float* outf, bf16_t* outb) {
    launch_gemm(stream, qsrc, D_, 0, 0, wT + 0 * DD, D_, 0, 0,
                nullptr, q_b, D_, 0, 0, MT_, D_, D_, bvec + 0 * D_, 0, 1, 1);
    launch_gemm(stream, kvsrc, D_, 0, 0, wT + 1 * DD, D_, 0, 0,
                nullptr, k_b, D_, 0, 0, MT_, D_, D_, bvec + 1 * D_, 0, 1, 1);
    launch_gemm(stream, kvsrc, D_, 0, 0, wT + 2 * DD, D_, 0, 0,
                nullptr, v_b, D_, 0, 0, MT_, D_, D_, bvec + 2 * D_, 0, 1, 1);
    k_transpose_heads<<<eg, 256, 0, stream>>>(v_b, vT_b);
    // scores[b,h] = Qh(S x HD) * Kh^T : Bt = Kh (S x HD)
    launch_gemm(stream,
                q_b, D_, (ll)S_ * D_, HD_,
                k_b, D_, (ll)S_ * D_, HD_,
                scores, nullptr, S_, (ll)H_ * S_ * S_, (ll)S_ * S_,
                S_, S_, HD_, nullptr, 0, B_ * H_, H_);
    k_softmax<<<B_ * H_ * S_, 128, 0, stream>>>(scores, att_b, y_in, causal, inv_sqrt_d);
    // ctx[b,h] = att(S x S) * Vh : Bt = Vh^T (HD x S); write into [B,S,D] bf16
    launch_gemm(stream,
                att_b, S_, (ll)H_ * S_ * S_, (ll)S_ * S_,
                vT_b, S_, (ll)H_ * HD_ * S_, (ll)HD_ * S_,
                nullptr, ctx_b, D_, (ll)S_ * D_, HD_,
                S_, HD_, S_, nullptr, 0, B_ * H_, H_);
    // O-projection -> fp32
    launch_gemm(stream, ctx_b, D_, 0, 0, wT + 3 * DD, D_, 0, 0,
                bufT, nullptr, D_, 0, 0, MT_, D_, D_, bvec + 3 * D_, 0, 1, 1);
    k_add_ln<<<MT_, 128, 0, stream>>>(bufT, resid, lnw_, lnb_, outf, outb);
  };

  auto ffn = [&](const bf16_t* inb, const float* resid,
                 const bf16_t* w1T, const float* b1, const bf16_t* w2T, const float* b2,
                 const float* lnw_, const float* lnb_, float* outf, bf16_t* outb) {
    launch_gemm(stream, inb, D_, 0, 0, w1T, D_, 0, 0,
                nullptr, ff1_b, FF_, 0, 0, MT_, FF_, D_, b1, 1, 1, 1);
    launch_gemm(stream, ff1_b, FF_, 0, 0, w2T, FF_, 0, 0,
                bufT, nullptr, D_, 0, 0, MT_, D_, FF_, b2, 0, 1, 1);
    k_add_ln<<<MT_, 128, 0, stream>>>(bufT, resid, lnw_, lnb_, outf, outb);
  };

  // ---- phase 1: source embedding + encoder (single block) ----
  k_pad_x<<<(MT_ * KPAD_ + 255) / 256, 256, 0, stream>>>(x_in, xpad);
  launch_gemm(stream, xpad, KPAD_, 0, 0, fcT, KPAD_, 0, 0,
              xe_f, nullptr, D_, 0, 0, MT_, D_, KPAD_, fc_b, 0, 1, 1);
  k_add_pos<<<eg, 256, 0, stream>>>(xe_f, pos_emb, xe_b);

  attention(xe_b, xe_b, encAT, eab, /*causal=*/0, xe_f, elnw, elnb, x1_f, x1_b);
  ffn(x1_b, x1_f, encF1T, ef1b, encF2T, ef2b, elnw + D_, elnb + D_, enc_f, enc_b);

  // ---- phase 2: target embedding ----
  k_embed_tgt<<<eg, 256, 0, stream>>>(y_in, emb_tab, h_f, h_b);

  // ---- phase 3: decoder stack ----
  for (int l = 0; l < L_; l++) {
    attention(h_b, h_b, decST + (size_t)l * 4 * DD, dsb + (size_t)l * 4 * D_,
              /*causal=*/1, h_f, dnw + (size_t)l * D_, dnb + (size_t)l * D_, qy_f, qy_b);
    attention(qy_b, enc_b, decCT + (size_t)l * 4 * DD, dcb + (size_t)l * 4 * D_,
              /*causal=*/0, qy_f, dlnw + (size_t)(l * 2) * D_, dlnb + (size_t)(l * 2) * D_,
              x1_f, x1_b);
    ffn(x1_b, x1_f, decF1T + (size_t)l * FF_ * D_, df1b + (size_t)l * FF_,
        decF2T + (size_t)l * D_ * FF_, df2b + (size_t)l * D_,
        dlnw + (size_t)(l * 2 + 1) * D_, dlnb + (size_t)(l * 2 + 1) * D_, h_f, h_b);
  }

  // ---- phase 4: output head ----
  launch_gemm(stream, h_b, D_, 0, 0, out1T, D_, 0, 0,
              nullptr, h1_b, D_, 0, 0, MT_, D_, D_, out1_b, 1, 1, 1);
  launch_gemm(stream, h1_b, D_, 0, 0, out2T, D_, 0, 0,
              logits, nullptr, V_, 0, 0, MT_, V_, D_, out2_b, 0, 1, 1);
}
